// CipherRNN_31344671326849
// MI455X (gfx1250) — matrix operand order repaired and checked by reading, not compiled, not run on hardware
//
#include <hip/hip_runtime.h>
#include <stdint.h>

// ---------------------------------------------------------------------------
// Problem dims
#define B_  64
#define S_  1024
#define V_  128
#define E_  512
#define H_  512
#define O_  256

// ---------------------------------------------------------------------------
// Vector types for WMMA / TDM
typedef __attribute__((ext_vector_type(16))) __bf16       v16bf;
typedef __attribute__((ext_vector_type(8)))  __bf16       v8bf;
typedef __attribute__((ext_vector_type(8)))  float        v8f;
typedef __attribute__((ext_vector_type(4)))  unsigned int v4u;
typedef __attribute__((ext_vector_type(8)))  int          v8i;
typedef __attribute__((ext_vector_type(4)))  int          v4i;

// ---------------------------------------------------------------------------
// Workspace layout (byte offsets, all 256-aligned)
#define OFF_BAR 0u                  // 2 counters (spread across 256B)
#define OFF_HB  256u                // h state ping-pong: 4 x 64x512 bf16 = 256KB
#define OFF_WB  262400u             // bf16 weights region (see element offsets)
#define OFF_XE  2752768u            // gathered embeddings bf16 [S*B, E] = 64MB
#define OFF_X0  69861632u           // precomputed layer0 preact f32 [S*B, H] = 128MB
#define OFF_HS  204079360u          // top-layer hidden bf16 [S*B, H] = 64MB

// bf16 weight region element offsets (u16 elements from OFF_WB)
#define W_XH0 0
#define W_XH1 262144
#define W_HH0 524288
#define W_HH1 786432
#define W_WHY 1048576
#define W_EMB 1179648
#define W_TOT 1245184

// ---------------------------------------------------------------------------
__device__ __forceinline__ __bf16 f2bf(float f) {
  union { float f; unsigned int u; } v; v.f = f;
  unsigned int u = v.u;
  unsigned short s = (unsigned short)((u + 0x7FFFu + ((u >> 16) & 1u)) >> 16);
  union { unsigned short s; __bf16 b; } o; o.s = s;
  return o.b;
}

// A fragment: 16x32 bf16 (row m = lane&15; K = kb..kb+7 and kb+16..kb+23, kb = lane<16?0:8)
__device__ __forceinline__ v16bf load_a_frag(const __bf16* A, int lda, int m0, int k0, int lane) {
  int r  = m0 + (lane & 15);
  int kb = (lane < 16) ? 0 : 8;
  const __bf16* p = A + (size_t)r * lda + (k0 + kb);
  v8bf lo = *(const v8bf*)(p);
  v8bf hi = *(const v8bf*)(p + 16);
  return __builtin_shufflevector(lo, hi, 0,1,2,3,4,5,6,7,8,9,10,11,12,13,14,15);
}

// B fragment: 32x16 bf16 from W[N,K] row-major (B[k,n] = W[n,k]):
// lane supplies col n = n0+(lane&15); K = kb..kb+7 and kb+16..kb+23, kb = k0 + (lane<16?0:8)
__device__ __forceinline__ v16bf load_b_frag(const __bf16* W, int ldw, int n0, int k0, int lane) {
  int n  = n0 + (lane & 15);
  int ko = k0 + ((lane < 16) ? 0 : 8);
  const __bf16* q = W + (size_t)n * ldw + ko;
  v8bf lo = *(const v8bf*)(q);
  v8bf hi = *(const v8bf*)(q + 16);
  return __builtin_shufflevector(lo, hi, 0,1,2,3,4,5,6,7,8,9,10,11,12,13,14,15);
}

__device__ __forceinline__ v8f wmma_bf16(v16bf a, v16bf b, v8f c) {
  return __builtin_amdgcn_wmma_f32_16x16x32_bf16(false, a, false, b, (short)0, c, false, false);
}

// ---------------------------------------------------------------------------
// Tensor Data Mover: 2D tile (dim1 rows x dim0 elems of 2B) global -> LDS
// This toolchain exposes the 6-arg builtin: (g0 v4u, g1 v8i, g2 v4i, g3 v4i, g4 v8i, cpol)
__device__ __forceinline__ void tdm_load_2d(unsigned int lds_off, const __bf16* gsrc,
                                            unsigned int dim0, unsigned int dim1,
                                            unsigned int stride0) {
  unsigned long long ga = (unsigned long long)(uintptr_t)gsrc;
  v4u g0;
  g0.x = 1u;                                              // count=1, user descriptor
  g0.y = lds_off;                                         // lds_addr (bytes)
  g0.z = (unsigned int)ga;                                // global_addr[31:0]
  g0.w = (unsigned int)((ga >> 32) & 0x1FFFFFFull) | (2u << 30);  // addr hi | type=2
  v8i g1;
  g1[0] = (int)(1u << 16);                                // wg_mask=0, data_size=1 (2B)
  g1[1] = (int)((dim0 & 0xFFFFu) << 16);                  // tensor_dim0[15:0]
  g1[2] = (int)(((dim0 >> 16) & 0xFFFFu) | ((dim1 & 0xFFFFu) << 16)); // dim0 hi | dim1 lo
  g1[3] = (int)(((dim1 >> 16) & 0xFFFFu) | ((dim0 & 0xFFFFu) << 16)); // dim1 hi | tile_dim0
  g1[4] = (int)(dim1 & 0xFFFFu);                          // tile_dim1 | tile_dim2=0
  g1[5] = (int)stride0;                                   // tensor_dim0_stride lo
  g1[6] = 0;
  g1[7] = 0;
  v4i g2 = (v4i)0;
  v4i g3 = (v4i)0;
  v8i g4 = (v8i)0;
  __builtin_amdgcn_tensor_load_to_lds(g0, g1, g2, g3, g4, 0);
}

// ---------------------------------------------------------------------------
// Device-wide barrier for the 16-WG persistent recurrence kernel.
// s_cluster_barrier is a NOP when not cluster-dispatched; atomic path is the
// portable fallback that is always correct for a fully-resident grid.
__device__ __forceinline__ void grid_barrier(unsigned int* bar) {
  __builtin_amdgcn_s_cluster_barrier();
  __threadfence();
  __syncthreads();
  if (threadIdx.x == 0) {
    unsigned int* cnt = bar;
    unsigned int* gen = bar + 32;   // separate 128B-apart line
    unsigned int g = __hip_atomic_load(gen, __ATOMIC_ACQUIRE, __HIP_MEMORY_SCOPE_AGENT);
    unsigned int prev = __hip_atomic_fetch_add(cnt, 1u, __ATOMIC_ACQ_REL, __HIP_MEMORY_SCOPE_AGENT);
    if (prev == 15u) {
      __hip_atomic_store(cnt, 0u, __ATOMIC_RELAXED, __HIP_MEMORY_SCOPE_AGENT);
      __hip_atomic_store(gen, g + 1u, __ATOMIC_RELEASE, __HIP_MEMORY_SCOPE_AGENT);
    } else {
      while (__hip_atomic_load(gen, __ATOMIC_ACQUIRE, __HIP_MEMORY_SCOPE_AGENT) == g)
        __builtin_amdgcn_s_sleep(1);
    }
  }
  __syncthreads();
  __threadfence();
}

// ---------------------------------------------------------------------------
// Prep: convert all weights f32->bf16, zero h state + barrier (re-init every
// launch so graph replay is deterministic).
__global__ void __launch_bounds__(256)
k_prep(const float* Wxh, const float* Whh, const float* Why, const float* emb,
       __bf16* wb, __bf16* hb, unsigned int* bar) {
  size_t tid    = (size_t)blockIdx.x * blockDim.x + threadIdx.x;
  size_t stride = (size_t)gridDim.x * blockDim.x;
  for (size_t i = tid; i < (size_t)W_TOT; i += stride) {
    float v;
    if (i < 524288)        v = Wxh[i];
    else if (i < 1048576)  v = Whh[i - 524288];
    else if (i < 1179648)  v = Why[i - 1048576];
    else                   v = emb[i - 1179648];
    wb[i] = f2bf(v);
  }
  for (size_t i = tid; i < 131072; i += stride) hb[i] = f2bf(0.0f);  // h0/h1 x2 bufs
  if (tid == 0) { bar[0] = 0u; bar[32] = 0u; }
}

// ---------------------------------------------------------------------------
// Gather: X_emb[(s*B+b), :] = emb_bf16[ids[b,s], :]   (time-major for the scan)
__global__ void __launch_bounds__(256)
k_gather(const int* __restrict__ ids, const __bf16* __restrict__ embb,
         __bf16* __restrict__ xe) {
  size_t i = (size_t)blockIdx.x * blockDim.x + threadIdx.x;  // S*B*E total
  int col = (int)(i & (E_ - 1));
  int row = (int)(i >> 9);
  int s = row >> 6, b = row & (B_ - 1);
  int id = ids[b * S_ + s];
  xe[i] = embb[(size_t)id * E_ + col];
}

// ---------------------------------------------------------------------------
// Parallel pre-GEMM: X0[S*B, H] = X_emb @ Wxh0^T + bh0  (one 16x16 tile / wave)
__global__ void __launch_bounds__(256)
k_gemm_x0(const __bf16* __restrict__ xe, const __bf16* __restrict__ wxh0,
          const float* __restrict__ bh, float* __restrict__ x0) {
  int lane = threadIdx.x & 31;
  int id   = blockIdx.x * 8 + (threadIdx.x >> 5);
  int m0 = (id >> 5) * 16;          // 4096 M tiles
  int n0 = (id & 31) * 16;          // 32 N tiles
  int n  = n0 + (lane & 15);
  int mo = m0 + ((lane < 16) ? 0 : 8);
  float bias = bh[n];
  v8f c;
  #pragma unroll
  for (int r = 0; r < 8; ++r) c[r] = bias;
  #pragma unroll 4
  for (int k0 = 0; k0 < E_; k0 += 32) {
    v16bf a = load_a_frag(xe, E_, m0, k0, lane);
    v16bf b = load_b_frag(wxh0, E_, n0, k0, lane);
    c = wmma_bf16(a, b, c);
  }
  #pragma unroll
  for (int r = 0; r < 8; ++r)
    x0[(size_t)(mo + r) * H_ + n] = c[r];
}

// ---------------------------------------------------------------------------
// Persistent sequential recurrence: 16 WGs x 8 waves = the 4x32 tile grid of
// the 64x512 state GEMMs. Recurrent weights live in LDS (TDM-loaded once).
__global__ void __launch_bounds__(256)
k_rnn(const float* __restrict__ x0, const float* __restrict__ bh, __bf16* hb,
      const __bf16* __restrict__ whh0, const __bf16* __restrict__ wxh1,
      const __bf16* __restrict__ whh1, __bf16* __restrict__ hseq,
      unsigned int* bar) {
  extern __shared__ __bf16 lw[];    // 3 x (32 rows x 512 cols) bf16 = 96KB
  const int wg   = blockIdx.x;      // 0..15 -> owns output cols [32*wg, 32*wg+32)
  const int lane = threadIdx.x & 31;
  const int wave = threadIdx.x >> 5;

  if (wave == 0) {  // one wave issues the three TDM tile loads
    tdm_load_2d(0u,     whh0 + (size_t)wg * 32 * H_, H_, 32, H_);
    tdm_load_2d(32768u, wxh1 + (size_t)wg * 32 * H_, H_, 32, H_);
    tdm_load_2d(65536u, whh1 + (size_t)wg * 32 * H_, H_, 32, H_);
    __builtin_amdgcn_s_wait_tensorcnt(0);
  }
  __syncthreads();

  const __bf16* lw_hh0 = lw;
  const __bf16* lw_xh1 = lw + 32 * H_;
  const __bf16* lw_hh1 = lw + 64 * H_;

  const int m0  = (wave & 3) * 16;             // 4 M tiles (B=64)
  const int ntl = wave >> 2;                   // 2 local N tiles
  const int n0g = wg * 32 + ntl * 16;          // global col base
  const int n0l = ntl * 16;                    // LDS-local W row base
  const int n   = n0g + (lane & 15);
  const int mo  = m0 + ((lane < 16) ? 0 : 8);
  const float bias1 = bh[H_ + n];

  __bf16* h0b[2] = { hb,          hb + 32768 };
  __bf16* h1b[2] = { hb + 65536,  hb + 98304 };

  for (int t = 0; t < S_; ++t) {
    const int p = t & 1;
    const __bf16* h0p = h0b[p];
    __bf16*       h0n = h0b[p ^ 1];
    const __bf16* h1p = h1b[p];
    __bf16*       h1n = h1b[p ^ 1];

    // ---- layer 0: h0 = tanh(X0[t] + h0_prev @ Whh0^T) ----
    const float* x0t = x0 + (size_t)t * B_ * H_;
    v8f c;
    #pragma unroll
    for (int r = 0; r < 8; ++r) c[r] = x0t[(size_t)(mo + r) * H_ + n];
    #pragma unroll 4
    for (int k0 = 0; k0 < H_; k0 += 32) {
      v16bf a = load_a_frag(h0p, H_, m0, k0, lane);
      v16bf b = load_b_frag(lw_hh0, H_, n0l, k0, lane);
      c = wmma_bf16(a, b, c);
    }
    #pragma unroll
    for (int r = 0; r < 8; ++r)
      h0n[(size_t)(mo + r) * H_ + n] = f2bf(tanhf(c[r]));

    grid_barrier(bar);

    // ---- layer 1: h1 = tanh(h0 @ Wxh1^T + h1_prev @ Whh1^T + bh1) ----
    #pragma unroll
    for (int r = 0; r < 8; ++r) c[r] = bias1;
    #pragma unroll 4
    for (int k0 = 0; k0 < H_; k0 += 32) {
      v16bf a = load_a_frag(h0n, H_, m0, k0, lane);
      v16bf b = load_b_frag(lw_xh1, H_, n0l, k0, lane);
      c = wmma_bf16(a, b, c);
    }
    #pragma unroll 4
    for (int k0 = 0; k0 < H_; k0 += 32) {
      v16bf a = load_a_frag(h1p, H_, m0, k0, lane);
      v16bf b = load_b_frag(lw_hh1, H_, n0l, k0, lane);
      c = wmma_bf16(a, b, c);
    }
    #pragma unroll
    for (int r = 0; r < 8; ++r) {
      __bf16 bv = f2bf(tanhf(c[r]));
      h1n[(size_t)(mo + r) * H_ + n] = bv;
      hseq[((size_t)t * B_ + (mo + r)) * H_ + n] = bv;   // top-layer emit
    }

    grid_barrier(bar);
  }
}

// ---------------------------------------------------------------------------
// Parallel post-GEMM: out[b, s, :] = hseq[(s*B+b), :] @ Why^T + by
__global__ void __launch_bounds__(256)
k_gemm_out(const __bf16* __restrict__ hs, const __bf16* __restrict__ why,
           const float* __restrict__ by, float* __restrict__ out) {
  int lane = threadIdx.x & 31;
  int id   = blockIdx.x * 8 + (threadIdx.x >> 5);
  int m0 = (id >> 4) * 16;          // 4096 M tiles
  int n0 = (id & 15) * 16;          // 16 N tiles (O=256)
  int n  = n0 + (lane & 15);
  int mo = m0 + ((lane < 16) ? 0 : 8);
  float bias = by[n];
  v8f c;
  #pragma unroll
  for (int r = 0; r < 8; ++r) c[r] = bias;
  #pragma unroll 4
  for (int k0 = 0; k0 < H_; k0 += 32) {
    v16bf a = load_a_frag(hs, H_, m0, k0, lane);
    v16bf b = load_b_frag(why, H_, n0, k0, lane);
    c = wmma_bf16(a, b, c);
  }
  #pragma unroll
  for (int r = 0; r < 8; ++r) {
    int rg = mo + r;
    int bb = rg & (B_ - 1);
    int ss = rg >> 6;
    out[((size_t)bb * S_ + ss) * O_ + n] = c[r];
  }
}

// ---------------------------------------------------------------------------
extern "C" void kernel_launch(void* const* d_in, const int* in_sizes, int n_in,
                              void* d_out, int out_size, void* d_ws, size_t ws_size,
                              hipStream_t stream) {
  const int*   ids = (const int*)d_in[0];
  const float* emb = (const float*)d_in[1];
  const float* Wxh = (const float*)d_in[2];
  const float* Whh = (const float*)d_in[3];
  const float* bh  = (const float*)d_in[4];
  const float* Why = (const float*)d_in[5];
  const float* by  = (const float*)d_in[6];
  float* out = (float*)d_out;

  char* ws = (char*)d_ws;
  unsigned int* bar = (unsigned int*)(ws + OFF_BAR);
  __bf16* hb = (__bf16*)(ws + OFF_HB);
  __bf16* wb = (__bf16*)(ws + OFF_WB);
  __bf16* xe = (__bf16*)(ws + OFF_XE);
  float*  x0 = (float*)(ws + OFF_X0);
  __bf16* hs = (__bf16*)(ws + OFF_HS);

  hipLaunchKernelGGL(k_prep,    dim3(1024),   dim3(256), 0, stream,
                     Wxh, Whh, Why, emb, wb, hb, bar);
  hipLaunchKernelGGL(k_gather,  dim3(131072), dim3(256), 0, stream,
                     ids, wb + W_EMB, xe);
  hipLaunchKernelGGL(k_gemm_x0, dim3(16384),  dim3(256), 0, stream,
                     xe, wb + W_XH0, bh, x0);
  hipLaunchKernelGGL(k_rnn,     dim3(16),     dim3(256), 98304, stream,
                     x0, bh, hb, wb + W_HH0, wb + W_XH1, wb + W_HH1, hs, bar);
  hipLaunchKernelGGL(k_gemm_out, dim3(8192),  dim3(256), 0, stream,
                     hs, wb + W_WHY, by, out);
}
